// TextEncoder_42580305773116
// MI455X (gfx1250) — compile-verified, matrix operand order checked by reading, not run on hardware
//
#include <hip/hip_runtime.h>

typedef _Float16 v16h __attribute__((ext_vector_type(16)));
typedef float    v8f  __attribute__((ext_vector_type(8)));

namespace {
constexpr int B = 8, T = 512, C = 512, H = 8, DK = 64, DFF = 2048, L = 6;
constexpr int NW = 9;                 // 2*W+1, W=4
constexpr int M = B * T;              // 4096 rows
constexpr int BCT = B * C * T;
constexpr float SQRT_C = 22.62741699796952f;  // sqrt(512)
constexpr int FLAG_MASK_IN = 1, FLAG_MASK_OUT = 2, FLAG_RELU = 4,
              FLAG_STORE_T = 8;
}

// A-fragment from fp32 memory (stride in elements).
// lane = 16*hi + idx; element j -> K offset hi*8+j, element j+8 -> 16+hi*8+j.
__device__ __forceinline__ v16h frag_ld(const float* __restrict__ p, int hi,
                                        int stride) {
  v16h r;
#pragma unroll
  for (int j = 0; j < 8; ++j) {
    r[j]     = (_Float16)p[(hi * 8 + j) * stride];
    r[j + 8] = (_Float16)p[(16 + hi * 8 + j) * stride];
  }
  return r;
}

// B-fragment from pre-converted f16 memory (contiguous -> 2x b128 per lane).
__device__ __forceinline__ v16h frag_ld_h(const _Float16* __restrict__ p,
                                          int hi) {
  v16h r;
#pragma unroll
  for (int j = 0; j < 8; ++j) {
    r[j]     = p[hi * 8 + j];
    r[j + 8] = p[16 + hi * 8 + j];
  }
  return r;
}

__device__ __forceinline__ float wave_sum(float v) {
#pragma unroll
  for (int off = 16; off > 0; off >>= 1) v += __shfl_xor(v, off, 32);
  return v;
}
__device__ __forceinline__ float wave_max(float v) {
#pragma unroll
  for (int off = 16; off > 0; off >>= 1) v = fmaxf(v, __shfl_xor(v, off, 32));
  return v;
}

// ---------------------------------------------------------------------------
// Weight convert fp32 [O, Cin, KK] -> f16 [KK, O, Cin]  (coalesced writes)
// ---------------------------------------------------------------------------
__global__ __launch_bounds__(256) void wcvt_kernel(
    const float* __restrict__ W, _Float16* __restrict__ Wh, int O, int Cin,
    int KK) {
  int idx = blockIdx.x * 256 + threadIdx.x;
  int total = O * Cin * KK;
  if (idx >= total) return;
  int kk = idx / (O * Cin);
  int rem = idx - kk * O * Cin;           // o*Cin + c
  Wh[idx] = (_Float16)W[(size_t)rem * KK + kk];
}

// ---------------------------------------------------------------------------
// Embedding: h[(b*T+t)*C+c] = emb[x[b,t],c]*sqrt(C)*mask ; x_emb output [B,C,T]
// ---------------------------------------------------------------------------
__global__ __launch_bounds__(256) void embed_kernel(
    const int* __restrict__ x, const int* __restrict__ lens,
    const float* __restrict__ emb, float* __restrict__ h,
    float* __restrict__ out_xemb) {
  int idx = blockIdx.x * 256 + threadIdx.x;  // over M*C
  int c = idx & (C - 1);
  int row = idx >> 9;  // / C
  int b = row >> 9;    // / T
  int t = row & (T - 1);
  float e = emb[x[row] * C + c] * SQRT_C;
  out_xemb[(b * C + c) * T + t] = e;  // x_emb is UNmasked
  h[idx] = (t < lens[b]) ? e : 0.f;
}

__global__ __launch_bounds__(256) void mask_rows_kernel(
    float* __restrict__ buf, const int* __restrict__ lens) {
  int idx = blockIdx.x * 256 + threadIdx.x;  // over M*C
  int row = idx >> 9;
  int b = row >> 9;
  int t = row & (T - 1);
  if (t >= lens[b]) buf[idx] = 0.f;
}

// ---------------------------------------------------------------------------
// WMMA GEMM / K-tap conv over time. One wave owns a 32(M) x 64(N) tile:
// 2 A-frags x 4 B-frags -> 8 WMMA per 32-wide K chunk (A reused 4x, B 2x).
//   Y[(b,t), o] = bias[o] + scale * sum_kk sum_c A[(b,t+kk-KK/2), c]*W[kk,o,c]
// A: [M, Cin] fp32.  Wh: [KK, Nout, Cin] f16 (pre-converted).
// ---------------------------------------------------------------------------
__global__ __launch_bounds__(256) void gemm_wmma_kernel(
    const float* __restrict__ A, const _Float16* __restrict__ Wh,
    const float* __restrict__ bias, float* __restrict__ Y, int Nout, int Cin,
    int KK, const int* __restrict__ lens, int flags, float scale) {
  const int lane = threadIdx.x & 31;
  const int wave = threadIdx.x >> 5;
  const int ntn = Nout >> 6;  // 64-col tiles
  const int tile = blockIdx.x * 8 + wave;
  const int tm = tile / ntn;
  const int tn = tile - tm * ntn;
  const int m = lane & 15, hi = lane >> 4;
  const int row0 = tm * 32 + m;        // A subtile 0 row
  const int row1 = row0 + 16;          // A subtile 1 row (same b: 32 | T)
  const int b0 = row0 >> 9, t0_ = row0 & (T - 1);
  const int t1_ = t0_ + 16;
  const int len = lens[b0];

  v8f acc[8] = {};                     // acc[sub*4 + n]
  for (int kk = 0; kk < KK; ++kk) {
    const int ts0 = t0_ + kk - (KK >> 1);
    const int ts1 = t1_ + kk - (KK >> 1);
    const bool mi = (flags & FLAG_MASK_IN) != 0;
    const bool v0 = (ts0 >= 0) && (ts0 < T) && (!mi || (ts0 < len));
    const bool v1 = (ts1 >= 0) && (ts1 < T) && (!mi || (ts1 < len));
    const _Float16* wbase = Wh + (size_t)kk * Nout * Cin;
    for (int k0 = 0; k0 < Cin; k0 += 32) {
      v16h a0 = {}, a1 = {};
      if (v0) a0 = frag_ld(A + (size_t)(b0 * T + ts0) * Cin + k0, hi, 1);
      if (v1) a1 = frag_ld(A + (size_t)(b0 * T + ts1) * Cin + k0, hi, 1);
#pragma unroll
      for (int n = 0; n < 4; ++n) {
        const _Float16* wp =
            wbase + (size_t)(tn * 64 + n * 16 + m) * Cin + k0;
        v16h bf = frag_ld_h(wp, hi);
        acc[n]     = __builtin_amdgcn_wmma_f32_16x16x32_f16(
            false, a0, false, bf, (short)0, acc[n], false, false);
        acc[4 + n] = __builtin_amdgcn_wmma_f32_16x16x32_f16(
            false, a1, false, bf, (short)0, acc[4 + n], false, false);
      }
    }
  }
#pragma unroll
  for (int sub = 0; sub < 2; ++sub) {
#pragma unroll
    for (int n = 0; n < 4; ++n) {
      const int ncol = tn * 64 + n * 16 + m;
      const float bb = bias[ncol];
      v8f a = acc[sub * 4 + n];
#pragma unroll
      for (int r = 0; r < 8; ++r) {
        int orow = tm * 32 + sub * 16 + r + 8 * hi;
        int ob = orow >> 9, ot = orow & (T - 1);
        float v = a[r] * scale + bb;
        if (flags & FLAG_RELU) v = fmaxf(v, 0.f);
        if ((flags & FLAG_MASK_OUT) && (ot >= lens[ob])) v = 0.f;
        if (flags & FLAG_STORE_T)
          Y[(ob * Nout + ncol) * T + ot] = v;   // [B, Nout, T] layout
        else
          Y[orow * Nout + ncol] = v;            // [M, Nout] layout
      }
    }
  }
}

// ---------------------------------------------------------------------------
// Fused attention for one (b, h, 16-row t-tile):
//   scores = QK^T (WMMA) + banded rel_k term; mask; softmax (LDS);
//   out = P V (WMMA, V pre-transposed to [B,C,T]) + banded rel_v term.
// ---------------------------------------------------------------------------
#define SCW 516
__global__ __launch_bounds__(256) void attn_kernel(
    const float* __restrict__ qb, const float* __restrict__ kb,
    const float* __restrict__ vt, const float* __restrict__ rk,
    const float* __restrict__ rv, float* __restrict__ ob,
    const int* __restrict__ lens) {
  __shared__ float sc[16][SCW];  // score / probability strip (16 x T)
  __shared__ float qt[16][DK];
  __shared__ float ot[16][DK];
  const int lane = threadIdx.x & 31, wave = threadIdx.x >> 5;
  const int m = lane & 15, hi = lane >> 4;
  const int blk = blockIdx.x;  // b*H*(T/16) blocks
  const int t16 = blk & 31;    // T/16 == 32
  const int bh = blk >> 5;
  const int h = bh & (H - 1);
  const int b = bh >> 3;  // H == 8
  const int t0 = t16 * 16;
  const int len = lens[b];
  const int hc = h * DK;

  for (int i = threadIdx.x; i < 16 * DK; i += 256) {
    int r = i >> 6, d = i & 63;
    qt[r][d] = qb[(b * T + t0 + r) * C + hc + d];
  }
  __syncthreads();

  // scores: 8 waves x 4 s-tiles of 16
#pragma unroll
  for (int i = 0; i < 4; ++i) {
    int sbase = (wave * 4 + i) * 16;
    v8f acc = {};
#pragma unroll
    for (int k0 = 0; k0 < DK; k0 += 32) {
      v16h af = frag_ld(&qt[m][k0], hi, 1);
      v16h bf = frag_ld(kb + (size_t)(b * T + sbase + m) * C + hc + k0, hi, 1);
      acc = __builtin_amdgcn_wmma_f32_16x16x32_f16(false, af, false, bf,
                                                   (short)0, acc, false, false);
    }
#pragma unroll
    for (int r = 0; r < 8; ++r) sc[r + 8 * hi][sbase + m] = acc[r];
  }
  __syncthreads();

  // banded relative-key term: scores[t, t+j-4] += q[t] . rk[j]
  if (threadIdx.x < 16 * NW) {
    int r = threadIdx.x / NW, j = threadIdx.x - r * NW;
    int s = t0 + r + j - 4;
    if (s >= 0 && s < T) {
      float a = 0.f;
      for (int d = 0; d < DK; ++d) a += qt[r][d] * rk[j * DK + d];
      sc[r][s] += a;
    }
  }
  __syncthreads();

  // mask (reference: where(mask==0, -1e4))
  for (int i = threadIdx.x; i < 16 * T; i += 256) {
    int r = i >> 9, s = i & (T - 1);
    if ((t0 + r) >= len || s >= len) sc[r][s] = -1e4f;
  }
  __syncthreads();

  // softmax: each wave owns 2 rows
#pragma unroll
  for (int rr = 0; rr < 2; ++rr) {
    int r = wave * 2 + rr;
    float mx = -3.4e38f;
    for (int s = lane; s < T; s += 32) mx = fmaxf(mx, sc[r][s]);
    mx = wave_max(mx);
    float sum = 0.f;
    for (int s = lane; s < T; s += 32) {
      float e = expf(sc[r][s] - mx);
      sc[r][s] = e;
      sum += e;
    }
    sum = wave_sum(sum);
    float inv = 1.f / sum;
    for (int s = lane; s < T; s += 32) sc[r][s] *= inv;
  }
  __syncthreads();

  // out = P V : waves 0..3 each own one 16-wide d-tile; V is [B,C,T]
  if (wave < 4) {
    int dbase = wave * 16;
    v8f acc = {};
    const float* vrow = vt + (size_t)(b * C + hc + dbase + m) * T;
    for (int k0 = 0; k0 < T; k0 += 32) {
      v16h af = frag_ld(&sc[m][k0], hi, 1);
      v16h bf = frag_ld(vrow + k0, hi, 1);
      acc = __builtin_amdgcn_wmma_f32_16x16x32_f16(false, af, false, bf,
                                                   (short)0, acc, false, false);
    }
#pragma unroll
    for (int r = 0; r < 8; ++r) ot[r + 8 * hi][dbase + m] = acc[r];
  }
  __syncthreads();

  // banded relative-value term + store
  for (int i = threadIdx.x; i < 16 * DK; i += 256) {
    int r = i >> 6, d = i & 63;
    int t = t0 + r;
    float v = ot[r][d];
#pragma unroll
    for (int j = 0; j < NW; ++j) {
      int s = t + j - 4;
      if (s >= 0 && s < T) v += sc[r][s] * rv[j * DK + d];
    }
    ob[(b * T + t) * C + hc + d] = v;
  }
}

// ---------------------------------------------------------------------------
// h = LayerNorm_C(h + y) * g + b  (one wave per row of C=512)
// ---------------------------------------------------------------------------
__global__ __launch_bounds__(256) void ln_residual_kernel(
    float* __restrict__ h, const float* __restrict__ y,
    const float* __restrict__ g, const float* __restrict__ bb) {
  const int lane = threadIdx.x & 31, wave = threadIdx.x >> 5;
  const int row = blockIdx.x * 8 + wave;
  float vals[16];
  float* hr = h + (size_t)row * C;
  const float* yr = y + (size_t)row * C;
  float s = 0.f;
#pragma unroll
  for (int i = 0; i < 16; ++i) {
    float v = hr[lane + 32 * i] + yr[lane + 32 * i];
    vals[i] = v;
    s += v;
  }
  s = wave_sum(s);
  float mean = s * (1.f / C);
  float vs = 0.f;
#pragma unroll
  for (int i = 0; i < 16; ++i) {
    float d = vals[i] - mean;
    vs += d * d;
  }
  vs = wave_sum(vs);
  float inv = rsqrtf(vs * (1.f / C) + 1e-5f);
#pragma unroll
  for (int i = 0; i < 16; ++i) {
    int c = lane + 32 * i;
    hr[c] = (vals[i] - mean) * inv * g[c] + bb[c];
  }
}

// ---------------------------------------------------------------------------
// Outputs: o_en = h*mask (as [B,C,T]), stats split into m / logs, x_mask.
// ---------------------------------------------------------------------------
__global__ __launch_bounds__(256) void finalize_kernel(
    const float* __restrict__ h, const float* __restrict__ S,
    const int* __restrict__ lens, float* __restrict__ out) {
  int idx = blockIdx.x * 256 + threadIdx.x;  // over M*C
  int c = idx & (C - 1);
  int row = idx >> 9;
  int b = row >> 9, t = row & (T - 1);
  float msk = (t < lens[b]) ? 1.f : 0.f;
  int o = (b * C + c) * T + t;
  out[BCT + o] = h[(size_t)row * C + c] * msk;  // o_en
  out[2 * BCT + o] = S[row * (2 * C) + c];      // m     (already masked)
  out[3 * BCT + o] = S[row * (2 * C) + C + c];  // logs  (already masked)
  if (c == 0) out[4 * BCT + b * T + t] = msk;   // x_mask [B,1,T]
}

// ---------------------------------------------------------------------------
extern "C" void kernel_launch(void* const* d_in, const int* in_sizes, int n_in,
                              void* d_out, int out_size, void* d_ws,
                              size_t ws_size, hipStream_t stream) {
  (void)in_sizes; (void)n_in; (void)out_size; (void)ws_size;
  const int*   x    = (const int*)d_in[0];
  const int*   lens = (const int*)d_in[1];
  const float* emb  = (const float*)d_in[2];
  const float* Wq   = (const float*)d_in[3];
  const float* bq   = (const float*)d_in[4];
  const float* Wk   = (const float*)d_in[5];
  const float* bk   = (const float*)d_in[6];
  const float* Wv   = (const float*)d_in[7];
  const float* bv   = (const float*)d_in[8];
  const float* Wo   = (const float*)d_in[9];
  const float* bo   = (const float*)d_in[10];
  const float* rlk  = (const float*)d_in[11];
  const float* rlv  = (const float*)d_in[12];
  const float* l1g  = (const float*)d_in[13];
  const float* l1b  = (const float*)d_in[14];
  const float* l2g  = (const float*)d_in[15];
  const float* l2b  = (const float*)d_in[16];
  const float* w1   = (const float*)d_in[17];
  const float* b1   = (const float*)d_in[18];
  const float* w2   = (const float*)d_in[19];
  const float* b2   = (const float*)d_in[20];
  const float* pw   = (const float*)d_in[21];
  const float* pb   = (const float*)d_in[22];
  float* out = (float*)d_out;

  float* ws   = (float*)d_ws;
  float* hbuf = ws;                    // [M, C]
  float* qbuf = hbuf + (size_t)M * C;  // [M, C]  (also attn-proj y)
  float* kbuf = qbuf + (size_t)M * C;  // [M, C]
  float* vbuf = kbuf + (size_t)M * C;  // [B, C, T] V^T  (also FFN y, [M, C])
  float* obuf = vbuf + (size_t)M * C;  // [M, C]
  float* fbuf = obuf + (size_t)M * C;  // [M, DFF]
  float* sbuf = fbuf;                  // [M, 2C] stats (reuse)

  // f16 weight staging area (layout [KK][O][Cin] per weight)
  _Float16* wqh = (_Float16*)(fbuf + (size_t)M * DFF);
  _Float16* wkh = wqh + (size_t)L * C * C;
  _Float16* wvh = wkh + (size_t)L * C * C;
  _Float16* woh = wvh + (size_t)L * C * C;
  _Float16* w1h = woh + (size_t)L * C * C;
  _Float16* w2h = w1h + (size_t)L * 3 * DFF * C;
  _Float16* pwh = w2h + (size_t)L * 3 * DFF * C;

  auto cvt = [&](const float* W, _Float16* Wh, int O, int Cin_, int KK_) {
    int total = O * Cin_ * KK_;
    wcvt_kernel<<<(total + 255) / 256, 256, 0, stream>>>(W, Wh, O, Cin_, KK_);
  };
  for (int l = 0; l < L; ++l) {
    cvt(Wq + (size_t)l * C * C, wqh + (size_t)l * C * C, C, C, 1);
    cvt(Wk + (size_t)l * C * C, wkh + (size_t)l * C * C, C, C, 1);
    cvt(Wv + (size_t)l * C * C, wvh + (size_t)l * C * C, C, C, 1);
    cvt(Wo + (size_t)l * C * C, woh + (size_t)l * C * C, C, C, 1);
    cvt(w1 + (size_t)l * DFF * C * 3, w1h + (size_t)l * 3 * DFF * C, DFF, C, 3);
    cvt(w2 + (size_t)l * C * DFF * 3, w2h + (size_t)l * 3 * C * DFF, C, DFF, 3);
  }
  cvt(pw, pwh, 2 * C, C, 1);

  const int ew_blocks = M * C / 256;  // 8192
  embed_kernel<<<ew_blocks, 256, 0, stream>>>(x, lens, emb, hbuf, out);

  // GEMM grids: tiles = (M/32)*(Nout/64), 8 waves (tiles) per block
  const int gC   = (M / 32) * (C / 64) / 8;        // 128
  const int gDFF = (M / 32) * (DFF / 64) / 8;      // 512
  const int g2C  = (M / 32) * (2 * C / 64) / 8;    // 256

  for (int l = 0; l < L; ++l) {
    mask_rows_kernel<<<ew_blocks, 256, 0, stream>>>(hbuf, lens);
    // Q/K/V projections (q folds 1/sqrt(DK); V stored transposed [B,C,T])
    gemm_wmma_kernel<<<gC, 256, 0, stream>>>(hbuf, wqh + (size_t)l * C * C,
                                             bq + l * C, qbuf, C, C, 1, lens,
                                             0, 0.125f);
    gemm_wmma_kernel<<<gC, 256, 0, stream>>>(hbuf, wkh + (size_t)l * C * C,
                                             bk + l * C, kbuf, C, C, 1, lens,
                                             0, 1.f);
    gemm_wmma_kernel<<<gC, 256, 0, stream>>>(hbuf, wvh + (size_t)l * C * C,
                                             bv + l * C, vbuf, C, C, 1, lens,
                                             FLAG_STORE_T, 1.f);
    attn_kernel<<<B * H * (T / 16), 256, 0, stream>>>(
        qbuf, kbuf, vbuf, rlk + l * NW * DK, rlv + l * NW * DK, obuf, lens);
    // output projection -> qbuf (q no longer needed)
    gemm_wmma_kernel<<<gC, 256, 0, stream>>>(obuf, woh + (size_t)l * C * C,
                                             bo + l * C, qbuf, C, C, 1, lens,
                                             0, 1.f);
    ln_residual_kernel<<<M / 8, 256, 0, stream>>>(hbuf, qbuf, l1g + l * C,
                                                  l1b + l * C);
    // FFN: conv K=3 up (relu), conv K=3 down (masked in+out) -> vbuf
    gemm_wmma_kernel<<<gDFF, 256, 0, stream>>>(
        hbuf, w1h + (size_t)l * 3 * DFF * C, b1 + l * DFF, fbuf, DFF, C, 3,
        lens, FLAG_MASK_IN | FLAG_RELU, 1.f);
    gemm_wmma_kernel<<<gC, 256, 0, stream>>>(
        fbuf, w2h + (size_t)l * 3 * C * DFF, b2 + l * C, vbuf, C, DFF, 3, lens,
        FLAG_MASK_IN | FLAG_MASK_OUT, 1.f);
    ln_residual_kernel<<<M / 8, 256, 0, stream>>>(hbuf, vbuf, l2g + l * C,
                                                  l2b + l * C);
  }

  // stats = proj(o_en) * mask, o_en = h * mask (mask folded via flags)
  gemm_wmma_kernel<<<g2C, 256, 0, stream>>>(
      hbuf, pwh, pb, sbuf, 2 * C, C, 1, lens, FLAG_MASK_IN | FLAG_MASK_OUT,
      1.f);
  finalize_kernel<<<ew_blocks, 256, 0, stream>>>(hbuf, sbuf, lens, out);
}
